// TopKMoE_45500883533877
// MI455X (gfx1250) — matrix operand order repaired and checked by reading, not compiled, not run on hardware
//
#include <hip/hip_runtime.h>

// ---------------------------------------------------------------------------
// Top-2 MoE for MI455X (gfx1250): compacted per-expert GEMMs using
// v_wmma_f32_16x16x32_bf16 (fp32 accumulate), fp32 gating/softmax.
// Double-buffered LDS pipeline (40 KB of the 320 KB/WGP) overlaps HBM/L2
// weight streaming with WMMA issue.
// ---------------------------------------------------------------------------

#define D_MODEL    1024
#define EXPERT_DIM 2048
#define NUM_EXP    16
#define N_TOK      4096          // SEQ_LEN * BATCH
#define N_SLOT     8192          // N_TOK * TOP_K
#define BM 128
#define BN 128
#define BK 32
#define BKP 40                   // padded LDS row (bf16 elems) -> 80B, 16B aligned

typedef __bf16 bf16_t;
typedef bf16_t v16bf __attribute__((ext_vector_type(16)));
typedef float  v8f   __attribute__((ext_vector_type(8)));

union Frag16 { v16bf v; uint4 q[2]; };

// ---------------- init: zero output + per-expert counters ------------------
__global__ void moe_init(float* __restrict__ out, int* __restrict__ counts) {
    int idx = blockIdx.x * blockDim.x + threadIdx.x;     // 4096*256 = 1,048,576 float4
    float4 z = {0.f, 0.f, 0.f, 0.f};
    ((float4*)out)[idx] = z;
    if (blockIdx.x == 0 && threadIdx.x < NUM_EXP) counts[threadIdx.x] = 0;
}

// ---------------- gating: logits, strict top-2, softmax --------------------
__global__ __launch_bounds__(256) void moe_gate(
    const float* __restrict__ x, const float* __restrict__ Wg,
    const float* __restrict__ bg, const float* __restrict__ bias,
    int* __restrict__ route_e, float* __restrict__ route_w, int* __restrict__ counts)
{
    __shared__ float lx[8][D_MODEL];       // 32 KB
    __shared__ float red[8][16][16];       //  8 KB
    __shared__ float lgt[8][16];
    const int tid = threadIdx.x;
    const int t0  = blockIdx.x * 8;

    for (int r = 0; r < 8; ++r) {
        float4 v = *(const float4*)(x + (size_t)(t0 + r) * D_MODEL + tid * 4);
        *(float4*)&lx[r][tid * 4] = v;
    }
    __syncthreads();

    const int e = tid & 15, c = tid >> 4;          // 16 experts x 16 chunks of 64
    float acc[8] = {0.f, 0.f, 0.f, 0.f, 0.f, 0.f, 0.f, 0.f};
    const float* wg = Wg + e;
    for (int d = c * 64; d < c * 64 + 64; ++d) {
        float w = wg[d * NUM_EXP];
        #pragma unroll
        for (int tk = 0; tk < 8; ++tk) acc[tk] += lx[tk][d] * w;
    }
    #pragma unroll
    for (int tk = 0; tk < 8; ++tk) red[tk][e][c] = acc[tk];
    __syncthreads();

    if (tid < 128) {
        int tk = tid >> 4, ee = tid & 15;
        float s = bg[ee] + bias[ee];
        #pragma unroll
        for (int cc = 0; cc < 16; ++cc) s += red[tk][ee][cc];
        lgt[tk][ee] = s;
    }
    __syncthreads();

    if (tid < 8) {
        const int tk = tid, token = t0 + tk;
        float v0 = -3.4e38f; int e0 = 0;
        for (int j = 0; j < 16; ++j) { float v = lgt[tk][j]; if (v > v0) { v0 = v; e0 = j; } }
        float v1 = -3.4e38f; int e1 = 0;
        for (int j = 0; j < 16; ++j) { if (j == e0) continue;
                                       float v = lgt[tk][j]; if (v > v1) { v1 = v; e1 = j; } }
        float ex = expf(v1 - v0);
        float w0 = 1.f / (1.f + ex);
        float w1 = ex * w0;
        route_e[token * 2 + 0] = e0;  route_w[token * 2 + 0] = w0;
        route_e[token * 2 + 1] = e1;  route_w[token * 2 + 1] = w1;
        atomicAdd(&counts[e0], 1);
        atomicAdd(&counts[e1], 1);
    }
}

// ---------------- exclusive scan over 16 counts ----------------------------
__global__ void moe_scan(const int* __restrict__ counts, int* __restrict__ base,
                         int* __restrict__ cursor) {
    if (threadIdx.x == 0) {
        int acc = 0;
        for (int e = 0; e < NUM_EXP; ++e) { base[e] = acc; cursor[e] = acc; acc += counts[e]; }
        base[NUM_EXP] = acc;
    }
}

// ---------------- compact slot assignment ----------------------------------
__global__ void moe_assign(const int* __restrict__ route_e, const float* __restrict__ route_w,
                           int* __restrict__ cursor, int* __restrict__ slot_token,
                           float* __restrict__ slot_w) {
    int i = blockIdx.x * 256 + threadIdx.x;
    if (i >= N_TOK) return;
    for (int k = 0; k < 2; ++k) {
        int   e = route_e[i * 2 + k];
        float w = route_w[i * 2 + k];
        int   s = atomicAdd(&cursor[e], 1);
        slot_token[s] = i;
        slot_w[s]     = w;
    }
}

// ---------------- gather tokens -> bf16 activation matrix ------------------
__global__ void moe_gather(const float* __restrict__ x, const int* __restrict__ slot_token,
                           bf16_t* __restrict__ Xg) {
    const int slot = blockIdx.x;
    const int tok  = slot_token[slot];
    const int d    = threadIdx.x * 4;
    float4 v = *(const float4*)(x + (size_t)tok * D_MODEL + d);
    union { bf16_t h[4]; uint2 u; } p;
    p.h[0] = (bf16_t)v.x; p.h[1] = (bf16_t)v.y; p.h[2] = (bf16_t)v.z; p.h[3] = (bf16_t)v.w;
    *(uint2*)(Xg + (size_t)slot * D_MODEL + d) = p.u;
}

// ---------------- ragged GEMM: 128x128 tile, K-step 32, 8 waves -----------
// MODE 0: H = relu(A*W + b1)  -> bf16 Hout (stride N)
// MODE 1: y = A*W + b2 ; atomicAdd(out[token], w_slot * y)
template <int MODE>
__global__ __launch_bounds__(256) void moe_gemm(
    const bf16_t* __restrict__ A, const float* __restrict__ W,
    const float* __restrict__ bvec, bf16_t* __restrict__ Hout, float* __restrict__ out,
    const int* __restrict__ base, const int* __restrict__ counts,
    const int* __restrict__ slot_token, const float* __restrict__ slot_w,
    int K, int N)
{
    const int e    = blockIdx.z;
    const int cnt  = counts[e];
    const int tm   = blockIdx.y;
    if (tm * BM >= cnt) return;                 // ragged: uniform per block
    const int sbase = base[e];
    const int n0    = blockIdx.x * BN;
    const int tid   = threadIdx.x;
    const int wave  = tid >> 5, lane = tid & 31;
    const int wm = wave & 3, wn = wave >> 2;    // 4 waves on M, 2 on N
    const int ln = lane & 15, hi = lane >> 4;

    __shared__ bf16_t lA[2][BM * BKP];          // 2 x 10 KB
    __shared__ bf16_t lB[2][BN * BKP];          // 2 x 10 KB (transposed W tile, bf16)

    v8f accv[2][4];
    #pragma unroll
    for (int i = 0; i < 2; ++i)
        #pragma unroll
        for (int j = 0; j < 4; ++j) accv[i][j] = (v8f){0, 0, 0, 0, 0, 0, 0, 0};

    const float* Wb = W + (size_t)e * K * N + n0;

    // stage tile [k0, k0+32) into LDS buffer `buf`
    auto stage = [&](int buf, int k0) {
        // A: 128 rows x 32 bf16, rows clamped into Xg range
        #pragma unroll
        for (int rep = 0; rep < 2; ++rep) {
            int q = tid + rep * 256;
            int row = q >> 2, qd = (q & 3) * 8;
            int s = sbase + tm * BM + row; if (s > N_SLOT - 1) s = N_SLOT - 1;
            uint4 v = *(const uint4*)(A + (size_t)s * K + k0 + qd);
            *(uint4*)&lA[buf][row * BKP + qd] = v;
        }
        // W: 32 fp32 rows x 128 cols -> bf16, transposed to lB[n][k]
        #pragma unroll
        for (int rep = 0; rep < 4; ++rep) {
            int q = tid + rep * 256;            // float4 units: 32 per row, 32 rows
            int kk = q >> 5, j4 = (q & 31) * 4;
            float4 f = *(const float4*)(Wb + (size_t)(k0 + kk) * N + j4);
            lB[buf][(j4 + 0) * BKP + kk] = (bf16_t)f.x;
            lB[buf][(j4 + 1) * BKP + kk] = (bf16_t)f.y;
            lB[buf][(j4 + 2) * BKP + kk] = (bf16_t)f.z;
            lB[buf][(j4 + 3) * BKP + kk] = (bf16_t)f.w;
        }
    };

    stage(0, 0);
    __syncthreads();

    int buf = 0;
    for (int k0 = 0; k0 < K; k0 += BK) {
        // prefetch next tile into the other buffer (overlaps with WMMA below)
        if (k0 + BK < K) stage(buf ^ 1, k0 + BK);

        Frag16 a[2], b[4];
        // A 16x32 bf16 layout: lanes 0-15 row M=ln, K 0-7 & 16-23; lanes 16-31 K 8-15 & 24-31
        #pragma unroll
        for (int mi = 0; mi < 2; ++mi) {
            int r = wm * 32 + mi * 16 + ln;
            a[mi].q[0] = *(const uint4*)&lA[buf][r * BKP + hi * 8];
            a[mi].q[1] = *(const uint4*)&lA[buf][r * BKP + 16 + hi * 8];
        }
        // B 32x16 bf16 layout: lanes 0-15 col N=ln K 0-7 & 16-23; lanes 16-31 K 8-15 & 24-31
        #pragma unroll
        for (int ni = 0; ni < 4; ++ni) {
            int cc = wn * 64 + ni * 16 + ln;
            b[ni].q[0] = *(const uint4*)&lB[buf][cc * BKP + hi * 8];
            b[ni].q[1] = *(const uint4*)&lB[buf][cc * BKP + 16 + hi * 8];
        }
        #pragma unroll
        for (int mi = 0; mi < 2; ++mi)
            #pragma unroll
            for (int ni = 0; ni < 4; ++ni)
                accv[mi][ni] = __builtin_amdgcn_wmma_f32_16x16x32_bf16(
                    false, a[mi].v, false, b[ni].v, (short)0, accv[mi][ni], false, false);

        __syncthreads();        // staged writes visible; old buffer free for reuse
        buf ^= 1;
    }

    // Epilogue. C/D layout: VGPR r -> M = mi*16 + hi*8 + r, N = ln.
    float bb[4];
    #pragma unroll
    for (int ni = 0; ni < 4; ++ni) bb[ni] = bvec[e * N + n0 + wn * 64 + ni * 16 + ln];

    #pragma unroll
    for (int mi = 0; mi < 2; ++mi) {
        #pragma unroll
        for (int r = 0; r < 8; ++r) {
            const int m = tm * BM + wm * 32 + mi * 16 + hi * 8 + r;
            if (m >= cnt) continue;                       // one predicate per row
            if (MODE == 0) {
                bf16_t* hrow = Hout + (size_t)(sbase + m) * N;
                #pragma unroll
                for (int ni = 0; ni < 4; ++ni) {
                    int ng = n0 + wn * 64 + ni * 16 + ln;
                    float v = accv[mi][ni][r] + bb[ni];
                    v = v > 0.f ? v : 0.f;
                    hrow[ng] = (bf16_t)v;
                }
            } else {
                const int   slot = sbase + m;
                const int   tok  = slot_token[slot];
                const float w    = slot_w[slot];
                float* orow = out + (size_t)tok * N;
                #pragma unroll
                for (int ni = 0; ni < 4; ++ni) {
                    int ng = n0 + wn * 64 + ni * 16 + ln;
                    atomicAdd(&orow[ng], w * (accv[mi][ni][r] + bb[ni]));
                }
            }
        }
    }
}

// ---------------------------------------------------------------------------
extern "C" void kernel_launch(void* const* d_in, const int* in_sizes, int n_in,
                              void* d_out, int out_size, void* d_ws, size_t ws_size,
                              hipStream_t stream) {
    (void)in_sizes; (void)n_in; (void)out_size; (void)ws_size;
    const float* x    = (const float*)d_in[0];
    const float* Wg   = (const float*)d_in[1];
    const float* bg   = (const float*)d_in[2];
    const float* bias = (const float*)d_in[3];
    const float* W1   = (const float*)d_in[4];
    const float* b1   = (const float*)d_in[5];
    const float* W2   = (const float*)d_in[6];
    const float* b2   = (const float*)d_in[7];
    float* out = (float*)d_out;

    size_t off = 0;
    char* wsb = (char*)d_ws;
    auto bump = [&](size_t bytes) -> void* {
        void* p = wsb + off;
        off += (bytes + 255) & ~(size_t)255;
        return p;
    };
    int*    counts     = (int*)bump(NUM_EXP * sizeof(int));
    int*    cursor     = (int*)bump(NUM_EXP * sizeof(int));
    int*    basep      = (int*)bump((NUM_EXP + 1) * sizeof(int));
    int*    route_e    = (int*)bump((size_t)N_SLOT * sizeof(int));
    float*  route_w    = (float*)bump((size_t)N_SLOT * sizeof(float));
    int*    slot_token = (int*)bump((size_t)N_SLOT * sizeof(int));
    float*  slot_w     = (float*)bump((size_t)N_SLOT * sizeof(float));
    bf16_t* Xg         = (bf16_t*)bump((size_t)N_SLOT * D_MODEL * sizeof(bf16_t));    // 16 MB
    bf16_t* H          = (bf16_t*)bump((size_t)N_SLOT * EXPERT_DIM * sizeof(bf16_t)); // 32 MB

    moe_init  <<<N_TOK, 256, 0, stream>>>(out, counts);
    moe_gate  <<<N_TOK / 8, 256, 0, stream>>>(x, Wg, bg, bias, route_e, route_w, counts);
    moe_scan  <<<1, 32, 0, stream>>>(counts, basep, cursor);
    moe_assign<<<N_TOK / 256, 256, 0, stream>>>(route_e, route_w, cursor, slot_token, slot_w);
    moe_gather<<<N_SLOT, 256, 0, stream>>>(x, slot_token, Xg);

    dim3 g1(EXPERT_DIM / BN, N_TOK / BM, NUM_EXP);   // (16, 32, 16)
    moe_gemm<0><<<g1, 256, 0, stream>>>(Xg, W1, b1, H, nullptr,
                                        basep, counts, slot_token, slot_w,
                                        D_MODEL, EXPERT_DIM);
    dim3 g2(D_MODEL / BN, N_TOK / BM, NUM_EXP);      // (8, 32, 16)
    moe_gemm<1><<<g2, 256, 0, stream>>>(H, W2, b2, nullptr, out,
                                        basep, counts, slot_token, slot_w,
                                        EXPERT_DIM, D_MODEL);
}
